// BAN_44341242364145
// MI455X (gfx1250) — matrix operand order repaired and checked by reading, not compiled, not run on hardware
//
#include <hip/hip_runtime.h>
#include <hip/hip_bf16.h>

// ---------------- types ----------------
typedef __bf16 bf16_t;
typedef __attribute__((ext_vector_type(16))) __bf16 v16bf;
typedef __attribute__((ext_vector_type(8)))  __bf16 v8bf;
typedef __attribute__((ext_vector_type(8)))  float  v8f;

static __device__ inline bf16_t f2bf(float f) {
  unsigned u = __builtin_bit_cast(unsigned, f);
  unsigned r = u + 0x7FFFu + ((u >> 16) & 1u);
  return __builtin_bit_cast(bf16_t, (unsigned short)(r >> 16));
}
static __device__ inline float bf2f(bf16_t h) {
  unsigned u = ((unsigned)__builtin_bit_cast(unsigned short, h)) << 16;
  return __builtin_bit_cast(float, u);
}

static __device__ inline v8f wmma_bf16(v16bf a, v16bf b, v8f c) {
  return __builtin_amdgcn_wmma_f32_16x16x32_bf16(false, a, false, b, (short)0, c,
                                                 false, false);
}

// ---- CDNA5 async global->LDS copy (ASYNCcnt-tracked, no VGPR round trip) ----
static __device__ inline unsigned lds_off(const void* p) {
  // flat address of an LDS object: addr[31:0] is the LDS byte offset (ISA 10.2)
  return (unsigned)(unsigned long long)p;
}
static __device__ inline void async_ld_b128(unsigned ldsaddr, const void* g) {
  asm volatile("global_load_async_to_lds_b128 %0, %1, off"
               :
               : "v"(ldsaddr), "v"(g)
               : "memory");
}
static __device__ inline void wait_async0() {
  asm volatile("s_wait_asynccnt 0" ::: "memory");
}

// A operand fragment (16x32 MxK, bf16). LDS tile row-major [rows][ldk] bf16.
static __device__ inline v16bf load_frag_a(const bf16_t* lds, int ldk, int row0,
                                           int k0, int lane) {
  int half = lane >> 4, m = lane & 15;
  const bf16_t* base = lds + (row0 + m) * ldk + k0 + half * 8;
  v8bf lo = *(const v8bf*)(base);
  v8bf hi = *(const v8bf*)(base + 16);
  return __builtin_shufflevector(lo, hi, 0, 1, 2, 3, 4, 5, 6, 7, 8, 9, 10, 11,
                                 12, 13, 14, 15);
}
// B operand fragment (32x16 KxN). LDS holds B transposed: [n][ldk] bf16.
static __device__ inline v16bf load_frag_b(const bf16_t* lds, int ldk, int n0,
                                           int k0, int lane) {
  int kh = lane >> 4, n = lane & 15;
  const bf16_t* base = lds + (n0 + n) * ldk + k0 + kh * 8;
  v8bf lo = *(const v8bf*)(base);
  v8bf hi = *(const v8bf*)(base + 16);
  return __builtin_shufflevector(lo, hi, 0, 1, 2, 3, 4, 5, 6, 7, 8, 9, 10, 11,
                                 12, 13, 14, 15);
}

// ---------------- pre-pass: conversions / weight transposes ----------------
__global__ __launch_bounds__(256) void conv_bf16(const float* __restrict__ in,
                                                 bf16_t* __restrict__ out,
                                                 int n) {
  int i = blockIdx.x * 256 + threadIdx.x;
  if (i < n) out[i] = f2bf(in[i]);
}
__global__ __launch_bounds__(256) void dual_copy(const float* __restrict__ src,
                                                 float* __restrict__ dstF,
                                                 bf16_t* __restrict__ dstB,
                                                 int n) {
  int i = blockIdx.x * 256 + threadIdx.x;
  if (i < n) {
    float v = src[i];
    dstF[i] = v;
    dstB[i] = f2bf(v);
  }
}
// out[N,K] bf16 = transpose(in[K,N] f32); batched over blockIdx.z
__global__ __launch_bounds__(256) void transpose_conv(
    const float* __restrict__ in, bf16_t* __restrict__ out, int K, int N) {
  __shared__ float tile[32][33];
  const float* inB = in + (size_t)blockIdx.z * K * N;
  bf16_t* outB = out + (size_t)blockIdx.z * K * N;
  int k0 = blockIdx.x * 32, n0 = blockIdx.y * 32;
  int tx = threadIdx.x & 31, ty = threadIdx.x >> 5;  // 32 x 8
#pragma unroll
  for (int j = 0; j < 32; j += 8)
    tile[ty + j][tx] = inB[(size_t)(k0 + ty + j) * N + n0 + tx];
  __syncthreads();
#pragma unroll
  for (int j = 0; j < 32; j += 8)
    outB[(size_t)(n0 + ty + j) * K + k0 + tx] = f2bf(tile[tx][ty + j]);
}

// ------- generic GEMM: out = act(A[MxK] @ Wt[NxK]^T + bias), all bf16 in -----
// M % 64 == 0, N % 128 == 0, K % 32 == 0.
__global__ __launch_bounds__(256) void gemm_bf16(
    const bf16_t* __restrict__ A, const bf16_t* __restrict__ Wt,
    const float* __restrict__ bias, float* __restrict__ outF,
    bf16_t* __restrict__ outB, int M, int N, int K, int relu) {
  __shared__ __align__(16) bf16_t sA[64 * 32];
  __shared__ __align__(16) bf16_t sB[128 * 32];  // [n][k]
  const int tid = threadIdx.x;
  const int wave = tid >> 5, lane = tid & 31;
  const int m0 = blockIdx.x * 64;
  const int n0 = blockIdx.y * 128;
  const int mtile = wave & 3;
  const int ngrp = wave >> 2;  // 0..1 -> 4 n-tiles each
  // staging coordinates (A: one 16B chunk/thread, B: two 16B chunks/thread)
  const int ar = tid >> 2, ac = (tid & 3) * 8;
  const int bn = tid >> 1, bc = (tid & 1) * 16;
  const unsigned ldsA = lds_off(&sA[ar * 32 + ac]);
  const unsigned ldsB = lds_off(&sB[bn * 32 + bc]);
  const bf16_t* gA = A + (size_t)(m0 + ar) * K + ac;
  const bf16_t* gB = Wt + (size_t)(n0 + bn) * K + bc;
  v8f acc[4] = {};
  for (int k0 = 0; k0 < K; k0 += 32) {
    async_ld_b128(ldsA, gA + k0);
    async_ld_b128(ldsB, gB + k0);
    async_ld_b128(ldsB + 16, gB + k0 + 8);
    wait_async0();
    __syncthreads();
    v16bf a = load_frag_a(sA, 32, mtile * 16, 0, lane);
#pragma unroll
    for (int j = 0; j < 4; ++j) {
      v16bf b = load_frag_b(sB, 32, (ngrp * 4 + j) * 16, 0, lane);
      acc[j] = wmma_bf16(a, b, acc[j]);
    }
    __syncthreads();
  }
#pragma unroll
  for (int j = 0; j < 4; ++j) {
    int nt = ngrp * 4 + j;
    int col = n0 + nt * 16 + (lane & 15);
    float bv = bias ? bias[col] : 0.0f;
#pragma unroll
    for (int r = 0; r < 8; ++r) {
      int row = m0 + mtile * 16 + r + ((lane >> 4) << 3);
      float v = acc[j][r] + bv;
      if (relu) v = fmaxf(v, 0.0f);
      size_t o = (size_t)row * N + col;
      if (outF) outF[o] = v;
      if (outB) outB[o] = f2bf(v);
    }
  }
}

// ---------------- v_mask ----------------
__global__ __launch_bounds__(256) void compute_vmask(
    const float* __restrict__ vrel, int* __restrict__ vmask) {
  int row = blockIdx.x;
  const float* p = vrel + (size_t)row * 2048;
  float s = 0.f;
  for (int i = threadIdx.x; i < 2048; i += 256) s += fabsf(p[i]);
  __shared__ float red[256];
  red[threadIdx.x] = s;
  __syncthreads();
  for (int off = 128; off > 0; off >>= 1) {
    if (threadIdx.x < off) red[threadIdx.x] += red[threadIdx.x + off];
    __syncthreads();
  }
  if (threadIdx.x == 0) vmask[row] = (red[0] == 0.0f) ? 1 : 0;
}

// ---------------- logits (WMMA) + masked softmax per (b,g) ----------------
__global__ __launch_bounds__(256) void logits_softmax(
    const bf16_t* __restrict__ vatt, const bf16_t* __restrict__ qatt,
    const float* __restrict__ h_mat, const float* __restrict__ h_bias,
    const int* __restrict__ vmask, float* __restrict__ att) {
  const int g = blockIdx.x, b = blockIdx.y;
  const int tid = threadIdx.x, wave = tid >> 5, lane = tid & 31;
  __shared__ __align__(16) bf16_t sA[112 * 32];
  __shared__ __align__(16) bf16_t sQ[32 * 32];  // [q][k]
  __shared__ float sLog[112 * 32];
  __shared__ float sRed[256];
  // async staging layout for A: 448 16-byte segments; each thread has <=2
  const int r0 = tid >> 2, c0 = (tid & 3) * 8;
  const int sr0 = (r0 < 100) ? r0 : 0;  // clamp pad rows (results masked later)
  const unsigned ldsA0 = lds_off(&sA[r0 * 32 + c0]);
  const bf16_t* gA0 = vatt + ((size_t)b * 100 + sr0) * 3072 + c0;
  const int seg1 = tid + 256;
  const int r1 = seg1 >> 2, c1 = (seg1 & 3) * 8;
  const int sr1 = (r1 < 100) ? r1 : 0;
  const bool has1 = (seg1 < 448);
  const unsigned ldsA1 = lds_off(&sA[(has1 ? r1 : 0) * 32 + (has1 ? c1 : 0)]);
  const bf16_t* gA1 = vatt + ((size_t)b * 100 + sr1) * 3072 + c1;
  v8f acc0 = {}, acc1 = {};
  for (int k0 = 0; k0 < 3072; k0 += 32) {
    async_ld_b128(ldsA0, gA0 + k0);
    if (has1) async_ld_b128(ldsA1, gA1 + k0);
    for (int idx = tid; idx < 32 * 32; idx += 256) {
      int r = idx >> 5, c = idx & 31;
      float q = (r < 20) ? bf2f(qatt[((size_t)b * 20 + r) * 3072 + k0 + c]) *
                               h_mat[g * 3072 + k0 + c]
                         : 0.0f;
      sQ[idx] = f2bf(q);
    }
    wait_async0();
    __syncthreads();
    if (wave < 7) {
      v16bf a = load_frag_a(sA, 32, wave * 16, 0, lane);
      v16bf b0 = load_frag_b(sQ, 32, 0, 0, lane);
      v16bf b1 = load_frag_b(sQ, 32, 16, 0, lane);
      acc0 = wmma_bf16(a, b0, acc0);
      acc1 = wmma_bf16(a, b1, acc1);
    }
    __syncthreads();
  }
  if (wave < 7) {
    float hb = h_bias[g];
#pragma unroll
    for (int nt = 0; nt < 2; ++nt) {
      v8f* ap = nt ? &acc1 : &acc0;
#pragma unroll
      for (int r = 0; r < 8; ++r) {
        int row = wave * 16 + r + ((lane >> 4) << 3);
        int col = nt * 16 + (lane & 15);
        bool bad = (row >= 100) || (col >= 20) ||
                   (row < 100 && vmask[b * 100 + row] != 0);
        sLog[row * 32 + col] = bad ? -__builtin_inff() : ((*ap)[r] + hb);
      }
    }
  }
  __syncthreads();
  float m = -__builtin_inff();
  for (int idx = tid; idx < 112 * 32; idx += 256) m = fmaxf(m, sLog[idx]);
  sRed[tid] = m;
  __syncthreads();
  for (int off = 128; off > 0; off >>= 1) {
    if (tid < off) sRed[tid] = fmaxf(sRed[tid], sRed[tid + off]);
    __syncthreads();
  }
  float mx = sRed[0];
  __syncthreads();
  float s = 0.f;
  for (int idx = tid; idx < 112 * 32; idx += 256) s += expf(sLog[idx] - mx);
  sRed[tid] = s;
  __syncthreads();
  for (int off = 128; off > 0; off >>= 1) {
    if (tid < off) sRed[tid] += sRed[tid + off];
    __syncthreads();
  }
  float inv = 1.0f / sRed[0];
  __syncthreads();
  for (int idx = tid; idx < 112 * 32; idx += 256) {
    int v = idx >> 5, q = idx & 31;
    if (v < 100 && q < 20)
      att[(((size_t)b * 8 + g) * 100 + v) * 20 + q] =
          expf(sLog[idx] - mx) * inv;
  }
}

// ---- fused: vg = relu(v_rel[b] @ b_v_W[g] + bias); u[b,g,q,:] = att^T @ vg ---
__global__ __launch_bounds__(256) void glimpse_v_pool(
    const bf16_t* __restrict__ vrelB, const bf16_t* __restrict__ bvWT,
    const float* __restrict__ bvB, const float* __restrict__ att,
    float* __restrict__ u) {
  const int ncol0 = blockIdx.x * 128;
  const int g = blockIdx.y;
  const int b = blockIdx.z;
  const int tid = threadIdx.x, wave = tid >> 5, lane = tid & 31;
  __shared__ __align__(16) bf16_t sA[112 * 32];
  __shared__ __align__(16) bf16_t sW[128 * 32];  // [n][k]
  __shared__ __align__(16) bf16_t sV[112 * 128];
  __shared__ float sAtt[2000];
  const bf16_t* WtG = bvWT + (size_t)g * 1024 * 2048;
  // A staging: 448 segments of 16B, <=2 per thread; pad rows clamp to row 0
  const int r0 = tid >> 2, c0 = (tid & 3) * 8;
  const int sr0 = (r0 < 100) ? r0 : 0;
  const unsigned ldsA0 = lds_off(&sA[r0 * 32 + c0]);
  const bf16_t* gA0 = vrelB + ((size_t)b * 100 + sr0) * 2048 + c0;
  const int seg1 = tid + 256;
  const int r1 = seg1 >> 2, c1 = (seg1 & 3) * 8;
  const int sr1 = (r1 < 100) ? r1 : 0;
  const bool has1 = (seg1 < 448);
  const unsigned ldsA1 = lds_off(&sA[(has1 ? r1 : 0) * 32 + (has1 ? c1 : 0)]);
  const bf16_t* gA1 = vrelB + ((size_t)b * 100 + sr1) * 2048 + c1;
  // W staging: two 16B chunks per thread
  const int bn = tid >> 1, bc = (tid & 1) * 16;
  const unsigned ldsW = lds_off(&sW[bn * 32 + bc]);
  const bf16_t* gW = WtG + (size_t)(ncol0 + bn) * 2048 + bc;
  v8f acc[7] = {};
  for (int k0 = 0; k0 < 2048; k0 += 32) {
    async_ld_b128(ldsA0, gA0 + k0);
    if (has1) async_ld_b128(ldsA1, gA1 + k0);
    async_ld_b128(ldsW, gW + k0);
    async_ld_b128(ldsW + 16, gW + k0 + 8);
    wait_async0();
    __syncthreads();
    v16bf bfr = load_frag_b(sW, 32, wave * 16, 0, lane);
#pragma unroll
    for (int mt = 0; mt < 7; ++mt) {
      v16bf a = load_frag_a(sA, 32, mt * 16, 0, lane);
      acc[mt] = wmma_bf16(a, bfr, acc[mt]);
    }
    __syncthreads();
  }
  {
    int col = wave * 16 + (lane & 15);
    float bv = bvB[g * 1024 + ncol0 + col];
#pragma unroll
    for (int mt = 0; mt < 7; ++mt)
#pragma unroll
      for (int r = 0; r < 8; ++r) {
        int row = mt * 16 + r + ((lane >> 4) << 3);
        sV[row * 128 + col] = f2bf(fmaxf(acc[mt][r] + bv, 0.0f));
      }
  }
  for (int idx = tid; idx < 2000; idx += 256)
    sAtt[idx] = att[(((size_t)b * 8 + g) * 100) * 20 + idx];
  __syncthreads();
  {
    int c = tid & 127;
    int q0 = (tid >> 7) * 10;
    float av[10];
#pragma unroll
    for (int j = 0; j < 10; ++j) av[j] = 0.f;
    for (int v = 0; v < 100; ++v) {
      float vv = bf2f(sV[v * 128 + c]);
#pragma unroll
      for (int j = 0; j < 10; ++j) av[j] += sAtt[v * 20 + q0 + j] * vv;
    }
#pragma unroll
    for (int j = 0; j < 10; ++j)
      u[(((size_t)b * 8 + g) * 20 + q0 + j) * 1024 + ncol0 + c] = av[j];
  }
}

// ---------------- small elementwise kernels ----------------
__global__ __launch_bounds__(256) void bemb_pool(const float* __restrict__ qg,
                                                 const float* __restrict__ u,
                                                 bf16_t* __restrict__ bembB,
                                                 int g) {
  int i = blockIdx.x * 256 + threadIdx.x;  // [0, 128*1024)
  int b = i >> 10, k = i & 1023;
  float s = 0.f;
  for (int q = 0; q < 20; ++q)
    s += qg[((size_t)b * 20 + q) * 1024 + k] *
         u[(((size_t)b * 8 + g) * 20 + q) * 1024 + k];
  bembB[i] = f2bf(s);
}
__global__ __launch_bounds__(256) void resid_add_dual(
    float* __restrict__ qcur, bf16_t* __restrict__ qcurB,
    const float* __restrict__ proj) {
  int i = blockIdx.x * 256 + threadIdx.x;  // [0, 128*20*1024)
  int b = i / (20 * 1024);
  int k = i & 1023;
  float v = qcur[i] + proj[b * 1024 + k];
  qcur[i] = v;
  qcurB[i] = f2bf(v);
}
__global__ __launch_bounds__(256) void joint_sum(const float* __restrict__ qcur,
                                                 float* __restrict__ out) {
  int i = blockIdx.x * 256 + threadIdx.x;  // [0, 128*1024)
  int b = i >> 10, k = i & 1023;
  float s = 0.f;
  for (int q = 0; q < 20; ++q) s += qcur[((size_t)b * 20 + q) * 1024 + k];
  out[i] = s;
}

// ---------------- host launcher ----------------
extern "C" void kernel_launch(void* const* d_in, const int* in_sizes, int n_in,
                              void* d_out, int out_size, void* d_ws,
                              size_t ws_size, hipStream_t stream) {
  (void)in_sizes; (void)n_in; (void)out_size; (void)ws_size;
  const float* v_rel = (const float*)d_in[0];
  const float* q_emb = (const float*)d_in[1];
  const float* Wva = (const float*)d_in[3];
  const float* bva = (const float*)d_in[4];
  const float* Wqa = (const float*)d_in[5];
  const float* bqa = (const float*)d_in[6];
  const float* h_mat = (const float*)d_in[7];
  const float* h_bias = (const float*)d_in[8];
  const float* bvW = (const float*)d_in[9];
  const float* bvB = (const float*)d_in[10];
  const float* bqW = (const float*)d_in[11];
  const float* bqB = (const float*)d_in[12];
  const float* qpW = (const float*)d_in[13];
  const float* qpB = (const float*)d_in[14];

  float* joint = (float*)d_out;          // [128*1024]
  float* att = (float*)d_out + 131072;   // [128*8*100*20]

  char* ws = (char*)d_ws;
  size_t off = 0;
  auto alloc = [&](size_t bytes) {
    void* p = ws + off;
    off += (bytes + 255) & ~(size_t)255;
    return p;
  };
  bf16_t* vrelB = (bf16_t*)alloc(12800ull * 2048 * 2);
  bf16_t* WvaT = (bf16_t*)alloc(2048ull * 3072 * 2);
  bf16_t* WqaT = (bf16_t*)alloc(1024ull * 3072 * 2);
  bf16_t* bvWT = (bf16_t*)alloc(8ull * 2048 * 1024 * 2);
  bf16_t* bqWT = (bf16_t*)alloc(8ull * 1024 * 1024 * 2);
  bf16_t* qpWT = (bf16_t*)alloc(8ull * 1024 * 1024 * 2);
  bf16_t* vattB = (bf16_t*)alloc(12800ull * 3072 * 2);
  bf16_t* qattB = (bf16_t*)alloc(2560ull * 3072 * 2);
  float* u = (float*)alloc(128ull * 8 * 20 * 1024 * 4);
  float* qcur = (float*)alloc(128ull * 20 * 1024 * 4);
  bf16_t* qcurB = (bf16_t*)alloc(128ull * 20 * 1024 * 2);
  float* qg = (float*)alloc(128ull * 20 * 1024 * 4);
  bf16_t* bembB = (bf16_t*)alloc(128ull * 1024 * 2);
  float* proj = (float*)alloc(128ull * 1024 * 4);
  int* vmask = (int*)alloc(12800 * 4);

  // ---- pre-pass: bf16 conversions + weight transposes ----
  dual_copy<<<10240, 256, 0, stream>>>(q_emb, qcur, qcurB, 128 * 20 * 1024);
  conv_bf16<<<102400, 256, 0, stream>>>(v_rel, vrelB, 12800 * 2048);
  transpose_conv<<<dim3(64, 96, 1), 256, 0, stream>>>(Wva, WvaT, 2048, 3072);
  transpose_conv<<<dim3(32, 96, 1), 256, 0, stream>>>(Wqa, WqaT, 1024, 3072);
  transpose_conv<<<dim3(64, 32, 8), 256, 0, stream>>>(bvW, bvWT, 2048, 1024);
  transpose_conv<<<dim3(32, 32, 8), 256, 0, stream>>>(bqW, bqWT, 1024, 1024);
  transpose_conv<<<dim3(32, 32, 8), 256, 0, stream>>>(qpW, qpWT, 1024, 1024);
  compute_vmask<<<12800, 256, 0, stream>>>(v_rel, vmask);

  // ---- attention ----
  gemm_bf16<<<dim3(200, 24), 256, 0, stream>>>(vrelB, WvaT, bva, nullptr,
                                               vattB, 12800, 3072, 2048, 1);
  gemm_bf16<<<dim3(40, 24), 256, 0, stream>>>(qcurB, WqaT, bqa, nullptr, qattB,
                                              2560, 3072, 1024, 1);
  logits_softmax<<<dim3(8, 128), 256, 0, stream>>>(vattB, qattB, h_mat, h_bias,
                                                   vmask, att);
  glimpse_v_pool<<<dim3(8, 8, 128), 256, 0, stream>>>(vrelB, bvWT, bvB, att, u);

  // ---- sequential glimpse updates ----
  for (int g = 0; g < 8; ++g) {
    gemm_bf16<<<dim3(40, 8), 256, 0, stream>>>(
        qcurB, bqWT + (size_t)g * 1024 * 1024, bqB + g * 1024, qg, nullptr,
        2560, 1024, 1024, 1);
    bemb_pool<<<512, 256, 0, stream>>>(qg, u, bembB, g);
    gemm_bf16<<<dim3(2, 8), 256, 0, stream>>>(
        bembB, qpWT + (size_t)g * 1024 * 1024, qpB + g * 1024, proj, nullptr,
        128, 1024, 1024, 0);
    resid_add_dual<<<10240, 256, 0, stream>>>(qcur, qcurB, proj);
  }
  joint_sum<<<512, 256, 0, stream>>>(qcur, joint);
}
